// MultiRegressionRNN_44487271252612
// MI455X (gfx1250) — matrix operand order repaired and checked by reading, not compiled, not run on hardware
//
#include <hip/hip_runtime.h>

// MI455X / gfx1250: two stacked tanh RNNCells + linear readout.
// One wave32 owns a 16-batch tile for all T=40 steps; each layer is a single
// v_wmma_f32_16x16x32_f16 with K-concatenated weights [W_in | W_hh] as the
// (constant) A operand, batch columns on N, and bias passed through C.
// State relayout D(f32 C-layout) -> B(f16) uses v_cvt_pk_rtz_f16_f32 plus
// ds_swizzle SWAPX16 lane-half swaps. tanh -> v_tanh_f32 when available.

#define BATCH_N 131072
#define T_N     40
#define IN_D    6
#define HID_N   16

typedef _Float16 v16h __attribute__((ext_vector_type(16)));
typedef float    v8f  __attribute__((ext_vector_type(8)));

union Mat { unsigned u[8]; v16h v; };

__device__ __forceinline__ unsigned pk_f16(float lo, float hi) {
  union { decltype(__builtin_amdgcn_cvt_pkrtz(0.f, 0.f)) h; unsigned u; } c;
  c.h = __builtin_amdgcn_cvt_pkrtz(lo, hi);
  return c.u;
}

// ds_swizzle group-of-32 SWAPX16: xor=0x10, or=0, and=0x1f -> imm 0x401f
#define HALF_SWAP(x) ((unsigned)__builtin_amdgcn_ds_swizzle((int)(x), 0x401f))

__device__ __forceinline__ float fast_tanh(float v) {
#if __has_builtin(__builtin_amdgcn_tanhf)
  return __builtin_amdgcn_tanhf(v);
#elif __has_builtin(__builtin_amdgcn_tanh_f32)
  return __builtin_amdgcn_tanh_f32(v);
#else
  float a = __builtin_fabsf(v);
  float e = __builtin_amdgcn_exp2f(a * -2.8853900817779268f); // exp(-2a)
  float r = (1.0f - e) * __builtin_amdgcn_rcpf(1.0f + e);
  return __builtin_copysignf(r, v);
#endif
}

__global__ __launch_bounds__(256) void rnn2_wmma_kernel(
    const float* __restrict__ X,
    const float* __restrict__ H1in, const float* __restrict__ H2in,
    const float* __restrict__ Wih1, const float* __restrict__ bih1,
    const float* __restrict__ Whh1, const float* __restrict__ bhh1,
    const float* __restrict__ Wih2, const float* __restrict__ bih2,
    const float* __restrict__ Whh2, const float* __restrict__ bhh2,
    const float* __restrict__ Wfc,  const float* __restrict__ bfc,
    float* __restrict__ out)
{
  const int lane = threadIdx.x & 31;
  const int wave = threadIdx.x >> 5;
  const int tile = blockIdx.x * 8 + wave;          // 8192 tiles of 16 batch rows
  const int base = tile * 16;
  const int bcol = base + (lane & 15);             // batch column this lane serves
  const bool lo  = (lane < 16);

  // ---- A operands (f16 A-layout: lane holds row M=lane&15; VGPR v, low half
  // K=2v / 16+2(v-4), high half +8) ----
  const int m     = lane & 15;
  const int khalf = lo ? 0 : 8;
  Mat A1, A2, Af;
#pragma unroll
  for (int v = 0; v < 8; ++v) {
    const int k0 = (v < 4) ? (2 * v + khalf) : (16 + 2 * (v - 4) + khalf);
    const int k1 = k0 + 1;
    // layer1 combined K: [0..5]=Wih1, [6..15]=0 pad, [16..31]=Whh1
    float w1a = (k0 < IN_D) ? Wih1[m * IN_D + k0]
              : (k0 >= 16 ? Whh1[m * HID_N + (k0 - 16)] : 0.f);
    float w1b = (k1 < IN_D) ? Wih1[m * IN_D + k1]
              : (k1 >= 16 ? Whh1[m * HID_N + (k1 - 16)] : 0.f);
    A1.u[v] = pk_f16(w1a, w1b);
    // layer2 combined K: [0..15]=Wih2, [16..31]=Whh2
    float w2a = (k0 < 16) ? Wih2[m * HID_N + k0] : Whh2[m * HID_N + (k0 - 16)];
    float w2b = (k1 < 16) ? Wih2[m * HID_N + k1] : Whh2[m * HID_N + (k1 - 16)];
    A2.u[v] = pk_f16(w2a, w2b);
    // readout: rows m<6 valid; K [0..15]=Wfc, [16..31]=0 pad
    float wfa = (m < IN_D && k0 < 16) ? Wfc[m * HID_N + k0] : 0.f;
    float wfb = (m < IN_D && k1 < 16) ? Wfc[m * HID_N + k1] : 0.f;
    Af.u[v] = pk_f16(wfa, wfb);
  }

  // ---- biases as C operand (f32 C-layout: VGPR v -> row v (lo half) / v+8) ----
  v8f c1, c2, cf;
#pragma unroll
  for (int v = 0; v < 8; ++v) {
    const int mr = v + (lo ? 0 : 8);
    c1[v] = bih1[mr] + bhh1[mr];
    c2[v] = bih2[mr] + bhh2[mr];
    cf[v] = (mr < IN_D) ? bfc[mr] : 0.f;
  }

  // ---- initial states, loaded straight into C-layout (2x b128 per lane) ----
  float h1f[8], h2f[8];
  {
    const float* p1 = H1in + (size_t)bcol * HID_N + (lo ? 0 : 8);
    const float* p2 = H2in + (size_t)bcol * HID_N + (lo ? 0 : 8);
    float4 a = *(const float4*)p1, b = *(const float4*)(p1 + 4);
    h1f[0]=a.x; h1f[1]=a.y; h1f[2]=a.z; h1f[3]=a.w;
    h1f[4]=b.x; h1f[5]=b.y; h1f[6]=b.z; h1f[7]=b.w;
    a = *(const float4*)p2; b = *(const float4*)(p2 + 4);
    h2f[0]=a.x; h2f[1]=a.y; h2f[2]=a.z; h2f[3]=a.w;
    h2f[4]=b.x; h2f[5]=b.y; h2f[6]=b.z; h2f[7]=b.w;
  }
  unsigned h1pk[4], h1sw[4], h2pk[4], h2sw[4];
#pragma unroll
  for (int j = 0; j < 4; ++j) {
    h1pk[j] = pk_f16(h1f[2*j], h1f[2*j+1]);  h1sw[j] = HALF_SWAP(h1pk[j]);
    h2pk[j] = pk_f16(h2f[2*j], h2f[2*j+1]);  h2sw[j] = HALF_SWAP(h2pk[j]);
  }

  const float* xrow = X   + (size_t)bcol * T_N * IN_D;
  float*       yrow = out + (size_t)bcol * T_N * IN_D;

  for (int t = 0; t < T_N; ++t) {
    // x_t for the 16 batch columns lives only in lanes 0-15 (3x b64 each)
    unsigned xp0 = 0, xp1 = 0, xp2 = 0;
    if (lo) {
      const float2* xp = (const float2*)(xrow + t * IN_D);
      float2 xa = xp[0], xb = xp[1], xc = xp[2];
      xp0 = pk_f16(xa.x, xa.y);
      xp1 = pk_f16(xb.x, xb.y);
      xp2 = pk_f16(xc.x, xc.y);
    }

    // B1 = [x (K0-5) ; pad ; h1 (K16-31)]
    Mat B;
    B.u[0] = lo ? xp0 : h1sw[0];
    B.u[1] = lo ? xp1 : h1sw[1];
    B.u[2] = lo ? xp2 : h1sw[2];
    B.u[3] = h1sw[3];                 // low half = K6,7 (A cols are 0)
    B.u[4] = h1pk[0];                 // low half = K16-23 = h1[0..7], high half = K24-31 = h1[8..15]
    B.u[5] = h1pk[1];
    B.u[6] = h1pk[2];
    B.u[7] = h1pk[3];

    v8f d = __builtin_amdgcn_wmma_f32_16x16x32_f16(
        false, A1.v, false, B.v, (short)0, c1, false, false);
#pragma unroll
    for (int i = 0; i < 8; ++i) h1f[i] = fast_tanh(d[i]);
#pragma unroll
    for (int j = 0; j < 4; ++j) {
      h1pk[j] = pk_f16(h1f[2*j], h1f[2*j+1]);
      h1sw[j] = HALF_SWAP(h1pk[j]);
    }

    // B2 = [h1_new (K0-15) ; h2_old (K16-31)]
    B.u[0] = h1pk[0];
    B.u[1] = h1pk[1];
    B.u[2] = h1pk[2];
    B.u[3] = h1pk[3];
    B.u[4] = h2pk[0];
    B.u[5] = h2pk[1];
    B.u[6] = h2pk[2];
    B.u[7] = h2pk[3];

    d = __builtin_amdgcn_wmma_f32_16x16x32_f16(
        false, A2.v, false, B.v, (short)0, c2, false, false);
#pragma unroll
    for (int i = 0; i < 8; ++i) h2f[i] = fast_tanh(d[i]);
#pragma unroll
    for (int j = 0; j < 4; ++j) {
      h2pk[j] = pk_f16(h2f[2*j], h2f[2*j+1]);
      h2sw[j] = HALF_SWAP(h2pk[j]);
    }

    // B_fc = [h2 (K0-15) ; don't-care (A cols 16-31 are 0)]
    B.u[0] = h2pk[0];  B.u[1] = h2pk[1];  B.u[2] = h2pk[2];  B.u[3] = h2pk[3];
    B.u[4] = h2sw[0];  B.u[5] = h2sw[1];  B.u[6] = h2sw[2];  B.u[7] = h2sw[3];

    d = __builtin_amdgcn_wmma_f32_16x16x32_f16(
        false, Af.v, false, B.v, (short)0, cf, false, false);

    // y rows 0..5 are in VGPRs 0..5, lanes 0-15 (batch = lane)
    if (lo) {
      float2* yp = (float2*)(yrow + t * IN_D);
      yp[0] = make_float2(d[0], d[1]);
      yp[1] = make_float2(d[2], d[3]);
      yp[2] = make_float2(d[4], d[5]);
    }
  }

  // ---- final states: each lane stores a contiguous 8-float chunk (2x b128) ----
  float* h1o = out + (size_t)BATCH_N * T_N * IN_D
             + (size_t)bcol * HID_N + (lo ? 0 : 8);
  float* h2o = h1o + (size_t)BATCH_N * HID_N;
  *(float4*)(h1o)     = make_float4(h1f[0], h1f[1], h1f[2], h1f[3]);
  *(float4*)(h1o + 4) = make_float4(h1f[4], h1f[5], h1f[6], h1f[7]);
  *(float4*)(h2o)     = make_float4(h2f[0], h2f[1], h2f[2], h2f[3]);
  *(float4*)(h2o + 4) = make_float4(h2f[4], h2f[5], h2f[6], h2f[7]);
}

extern "C" void kernel_launch(void* const* d_in, const int* in_sizes, int n_in,
                              void* d_out, int out_size, void* d_ws, size_t ws_size,
                              hipStream_t stream) {
  const float* X    = (const float*)d_in[0];
  const float* H1   = (const float*)d_in[1];
  const float* H2   = (const float*)d_in[2];
  const float* Wih1 = (const float*)d_in[3];
  const float* bih1 = (const float*)d_in[4];
  const float* Whh1 = (const float*)d_in[5];
  const float* bhh1 = (const float*)d_in[6];
  const float* Wih2 = (const float*)d_in[7];
  const float* bih2 = (const float*)d_in[8];
  const float* Whh2 = (const float*)d_in[9];
  const float* bhh2 = (const float*)d_in[10];
  const float* Wfc  = (const float*)d_in[11];
  const float* bfc  = (const float*)d_in[12];
  float* out = (float*)d_out;

  // 8192 16-batch tiles, 8 waves (tiles) per 256-thread block -> 1024 blocks
  dim3 grid(BATCH_N / 16 / 8);
  dim3 block(256);
  rnn2_wmma_kernel<<<grid, block, 0, stream>>>(
      X, H1, H2, Wih1, bih1, Whh1, bhh1, Wih2, bih2, Whh2, bhh2, Wfc, bfc, out);
}
